// LatticeEncoderLayer_59184649339445
// MI455X (gfx1250) — matrix operand order repaired and checked by reading, not compiled, not run on hardware
//
#include <hip/hip_runtime.h>
#include <hip/hip_bf16.h>

// Problem constants (match reference)
#define BB 8
#define SS 4096
#define DD 512
#define FF 4
#define MTOT (BB * SS)       // 32768 rows
#define K2D (2 * DD)         // 1024
#define ROWS_PER_BLK 32
#define LDS_STRIDE 1032      // 1024 + 8 halfs padding (bank spread)

typedef __attribute__((ext_vector_type(16))) _Float16 v16h;
typedef __attribute__((ext_vector_type(8)))  _Float16 v8h;
typedef __attribute__((ext_vector_type(4)))  _Float16 v4h;
typedef __attribute__((ext_vector_type(8)))  float    v8f;

// ---------------------------------------------------------------------------
// Kernel 1: W_out [2D, D] f32 (row-major, W[k][n]) -> Wt [D, 2D] f16 (Wt[n][k])
// ---------------------------------------------------------------------------
__global__ void lel_transpose_w(const float* __restrict__ W,
                                _Float16* __restrict__ Wt) {
    int idx = blockIdx.x * blockDim.x + threadIdx.x;  // 512*1024 total
    if (idx < DD * K2D) {
        int n = idx >> 10;          // 0..511
        int k = idx & (K2D - 1);    // 0..1023
        Wt[idx] = (_Float16)W[k * DD + n];
    }
}

// ---------------------------------------------------------------------------
// Kernel 2: fused attention + gated GEMM epilogue.
// grid.x = MTOT/32 = 1024 blocks, 512 threads (16 waves of 32).
// ---------------------------------------------------------------------------
__global__ void lel_fused(const float* __restrict__ inputs,   // [M, D]
                          const float* __restrict__ values,   // [M, F*D]
                          const float* __restrict__ w_score,  // [D]
                          const float* __restrict__ b_score,  // [1]
                          const int*   __restrict__ n_ptr,    // [1]
                          const _Float16* __restrict__ Wt,    // [D, 2D] f16
                          float* __restrict__ out) {          // [M, D]
    extern __shared__ char smemraw[];
    _Float16* As   = (_Float16*)smemraw;                       // [32][LDS_STRIDE]
    float*    gateS = (float*)(smemraw + ROWS_PER_BLK * LDS_STRIDE * 2);

    const int lane  = threadIdx.x & 31;
    const int wv    = threadIdx.x >> 5;    // 0..15
    const int mlane = lane & 15;
    const int ksel  = lane >> 4;           // 0 or 1
    const long m0   = (long)blockIdx.x * ROWS_PER_BLK;
    const int  nn   = *n_ptr;
    const float bsc = *b_score;

    // ---------------- Phase 1: per-row attention, build A-tile in LDS -------
    for (int rr = 0; rr < 2; ++rr) {
        const int  rloc = wv * 2 + rr;           // 0..31
        const long m    = m0 + rloc;
        const float* qp = inputs + m * DD;
        const float* mp = values + m * (FF * DD);

        float s0 = 0.f, s1 = 0.f, s2 = 0.f, s3 = 0.f, sg = 0.f;
        for (int c = 0; c < 4; ++c) {
            const int d = c * 128 + lane * 4;
            float4 q4 = *(const float4*)(qp + d);
            float4 w4 = *(const float4*)(w_score + d);
            float4 a4 = *(const float4*)(mp + 0 * DD + d);
            float4 b4 = *(const float4*)(mp + 1 * DD + d);
            float4 c4 = *(const float4*)(mp + 2 * DD + d);
            float4 d4 = *(const float4*)(mp + 3 * DD + d);
            s0 += q4.x * a4.x + q4.y * a4.y + q4.z * a4.z + q4.w * a4.w;
            s1 += q4.x * b4.x + q4.y * b4.y + q4.z * b4.z + q4.w * b4.w;
            s2 += q4.x * c4.x + q4.y * c4.y + q4.z * c4.z + q4.w * c4.w;
            s3 += q4.x * d4.x + q4.y * d4.y + q4.z * d4.z + q4.w * d4.w;
            sg += q4.x * w4.x + q4.y * w4.y + q4.z * w4.z + q4.w * w4.w;
        }
        // full butterfly reduce (wave32) -> every lane has totals
        for (int off = 16; off >= 1; off >>= 1) {
            s0 += __shfl_xor(s0, off, 32);
            s1 += __shfl_xor(s1, off, 32);
            s2 += __shfl_xor(s2, off, 32);
            s3 += __shfl_xor(s3, off, 32);
            sg += __shfl_xor(sg, off, 32);
        }
        float sc[FF] = {s0, s1, s2, s3};
        #pragma unroll
        for (int f = 0; f < FF; ++f)
            if (f >= nn) sc[f] = -1e9f;
        float mx = fmaxf(fmaxf(sc[0], sc[1]), fmaxf(sc[2], sc[3]));
        float e0 = __expf(sc[0] - mx), e1 = __expf(sc[1] - mx);
        float e2 = __expf(sc[2] - mx), e3 = __expf(sc[3] - mx);
        float inv = 1.f / (e0 + e1 + e2 + e3);
        const float a0 = e0 * inv, a1 = e1 * inv, a2 = e2 * inv, a3 = e3 * inv;
        const float gate = 1.f / (1.f + __expf(-(sg + bsc)));
        if (lane == 0) gateS[rloc] = gate;

        // pass 2: ctx + q -> f16 LDS tile. cat row = [ctx(0..511) | q(512..1023)]
        _Float16* arow = As + rloc * LDS_STRIDE;
        for (int c = 0; c < 4; ++c) {
            const int d = c * 128 + lane * 4;
            float4 q4 = *(const float4*)(qp + d);
            float4 a4 = *(const float4*)(mp + 0 * DD + d);
            float4 b4 = *(const float4*)(mp + 1 * DD + d);
            float4 c4 = *(const float4*)(mp + 2 * DD + d);
            float4 d4 = *(const float4*)(mp + 3 * DD + d);
            float cx = a0 * a4.x + a1 * b4.x + a2 * c4.x + a3 * d4.x;
            float cy = a0 * a4.y + a1 * b4.y + a2 * c4.y + a3 * d4.y;
            float cz = a0 * a4.z + a1 * b4.z + a2 * c4.z + a3 * d4.z;
            float cw = a0 * a4.w + a1 * b4.w + a2 * c4.w + a3 * d4.w;
            v4h ctxh = {(_Float16)cx, (_Float16)cy, (_Float16)cz, (_Float16)cw};
            v4h qh   = {(_Float16)q4.x, (_Float16)q4.y, (_Float16)q4.z, (_Float16)q4.w};
            *(v4h*)(arow + d)      = ctxh;
            *(v4h*)(arow + DD + d) = qh;
        }
    }
    __syncthreads();

    // ---------------- Phase 2: WMMA GEMM over the tile ----------------------
    // wave wv owns N-tiles at columns n0a = wv*16 and n0b = (wv+16)*16,
    // and both M-tiles (rows 0-15 and 16-31).
    const int n0a = wv * 16;
    const int n0b = (wv + 16) * 16;

    v8f acc00 = {}, acc01 = {}, acc10 = {}, acc11 = {};

    const _Float16* rowA0 = As + mlane * LDS_STRIDE + ksel * 16;         // M-tile 0
    const _Float16* rowA1 = As + (16 + mlane) * LDS_STRIDE + ksel * 16;  // M-tile 1
    const _Float16* wtA = Wt + (long)(n0a + mlane) * K2D + ksel * 16;
    const _Float16* wtB = Wt + (long)(n0b + mlane) * K2D + ksel * 16;

    for (int k0 = 0; k0 < K2D; k0 += 32) {
        // A frags: contiguous K (elems 0-15 = K base..base+15, base = k0 + 16*ksel)
        v8h a0lo = *(const v8h*)(rowA0 + k0);
        v8h a0hi = *(const v8h*)(rowA0 + k0 + 8);
        v8h a1lo = *(const v8h*)(rowA1 + k0);
        v8h a1hi = *(const v8h*)(rowA1 + k0 + 8);
        v16h a0 = __builtin_shufflevector(a0lo, a0hi, 0,1,2,3,4,5,6,7,8,9,10,11,12,13,14,15);
        v16h a1 = __builtin_shufflevector(a1lo, a1hi, 0,1,2,3,4,5,6,7,8,9,10,11,12,13,14,15);
        // B frags: lanes 0-15 carry K=k0..k0+15, lanes 16-31 K=k0+16..k0+31
        v16h b0 = *(const v16h*)(wtA + k0);
        v16h b1 = *(const v16h*)(wtB + k0);

        acc00 = __builtin_amdgcn_wmma_f32_16x16x32_f16(false, a0, false, b0, (short)0, acc00, false, false);
        acc01 = __builtin_amdgcn_wmma_f32_16x16x32_f16(false, a0, false, b1, (short)0, acc01, false, false);
        acc10 = __builtin_amdgcn_wmma_f32_16x16x32_f16(false, a1, false, b0, (short)0, acc10, false, false);
        acc11 = __builtin_amdgcn_wmma_f32_16x16x32_f16(false, a1, false, b1, (short)0, acc11, false, false);
    }

    // ---------------- Epilogue: tanh * gate + residual ----------------------
    const int rowAdd = (lane < 16) ? 0 : 8;  // C/D layout: VGPR e -> M=e (+8 hi half)
    #pragma unroll
    for (int e = 0; e < 8; ++e) {
        {   // acc00: M-tile 0, N-tile a
            int rloc = e + rowAdd;
            long m = m0 + rloc; int col = n0a + mlane;
            out[m * DD + col] = tanhf(acc00[e]) * gateS[rloc] + inputs[m * DD + col];
        }
        {   // acc01: M-tile 0, N-tile b
            int rloc = e + rowAdd;
            long m = m0 + rloc; int col = n0b + mlane;
            out[m * DD + col] = tanhf(acc01[e]) * gateS[rloc] + inputs[m * DD + col];
        }
        {   // acc10: M-tile 1, N-tile a
            int rloc = 16 + e + rowAdd;
            long m = m0 + rloc; int col = n0a + mlane;
            out[m * DD + col] = tanhf(acc10[e]) * gateS[rloc] + inputs[m * DD + col];
        }
        {   // acc11: M-tile 1, N-tile b
            int rloc = 16 + e + rowAdd;
            long m = m0 + rloc; int col = n0b + mlane;
            out[m * DD + col] = tanhf(acc11[e]) * gateS[rloc] + inputs[m * DD + col];
        }
    }
}

// ---------------------------------------------------------------------------
extern "C" void kernel_launch(void* const* d_in, const int* in_sizes, int n_in,
                              void* d_out, int out_size, void* d_ws, size_t ws_size,
                              hipStream_t stream) {
    const float* inputs  = (const float*)d_in[0];   // [B,S,D]
    const float* values  = (const float*)d_in[1];   // [B,S,F*D]
    const float* W_out   = (const float*)d_in[2];   // [2D, D]
    const float* w_score = (const float*)d_in[3];   // [D,1]
    const float* b_score = (const float*)d_in[4];   // [1]
    const int*   n_fac   = (const int*)d_in[5];     // [1]
    float* out = (float*)d_out;

    _Float16* Wt = (_Float16*)d_ws;                 // needs 512*1024*2 = 1 MB

    // Prep: transpose+convert W_out into f16 [D][2D]
    lel_transpose_w<<<(DD * K2D + 511) / 512, 512, 0, stream>>>(W_out, Wt);

    // Fused kernel: 1024 blocks x 512 threads, 66176 B dynamic LDS
    const size_t lds = (size_t)ROWS_PER_BLK * LDS_STRIDE * 2 + ROWS_PER_BLK * 4;
    lel_fused<<<MTOT / ROWS_PER_BLK, 512, lds, stream>>>(
        inputs, values, w_score, b_score, n_fac, Wt, out);
}
